// LowRankMaskedSynapse_26250840113208
// MI455X (gfx1250) — matrix operand order repaired and checked by reading, not compiled, not run on hardware
//
#include <hip/hip_runtime.h>

#define B_DIM 512
#define N_DIM 16384
#define R_DIM 128
#define KSPLIT 16          // k-split for gemm_pre (512 waves)
#define NT_OUT 4           // n-tiles per wave in gemm_out

typedef __attribute__((ext_vector_type(16))) __bf16 v16bf;
typedef __attribute__((ext_vector_type(8)))  float v8f;

// ---------------- prep kernels ----------------

__global__ void zero_ws_kernel(float* __restrict__ ws, int nwords) {
    int i = blockIdx.x * blockDim.x + threadIdx.x;
    if (i < nwords) ws[i] = 0.0f;
}

__global__ void build_mask_kernel(const int* __restrict__ indices,
                                  unsigned* __restrict__ mask, int nnz) {
    int i = blockIdx.x * blockDim.x + threadIdx.x;
    if (i < nnz) {
        int j = indices[i];
        if (j >= 0 && j < N_DIM) mask[j] = 1u;   // all writers store 1: no atomic needed
    }
}

// V (f32, [N,R]) -> Vbf (bf16, same layout); coalesced both ways
__global__ void cvt_V_kernel(const float* __restrict__ V, __bf16* __restrict__ Vb, int n) {
    int i = blockIdx.x * blockDim.x + threadIdx.x;
    if (i < n) Vb[i] = (__bf16)V[i];
}

// U (f32, [K=16384, R=128]) -> Ut (bf16, [R, K]) via LDS tile transpose.
// Block: 256 threads, tile = 256 k x 16 n. Writes are 512B-contiguous per n-row.
#define TK 256
__global__ void __launch_bounds__(256) transpose_U_kernel(const float* __restrict__ U,
                                                          __bf16* __restrict__ Ut) {
    __shared__ __bf16 tile[TK][17];                 // padded to dodge bank conflicts
    const int k0 = blockIdx.x * TK;
    const int n0 = blockIdx.y * 16;
    const int t  = threadIdx.x;
    const int tn = t & 15;
    const int tk = t >> 4;                          // 0..15
    #pragma unroll 4
    for (int c = 0; c < TK; c += 16) {
        const int k = k0 + c + tk;
        tile[c + tk][tn] = (__bf16)U[(size_t)k * R_DIM + n0 + tn];
    }
    __syncthreads();
    #pragma unroll 4
    for (int nn = 0; nn < 16; ++nn)
        Ut[(size_t)(n0 + nn) * N_DIM + k0 + t] = tile[t][nn];
}

// ---------------- pre = x @ U  ([512,16384] @ [16384,128]) ----------------
// One wave per (16-row block, k-slice). Each wave computes the full 16x128
// output strip: A fragment (from x) loaded once per k-step, 8 B fragments
// (contiguous bf16 from Ut) + 8 WMMAs. f32 atomics reduce the KSPLIT partials.

__global__ void __launch_bounds__(32) gemm_pre_kernel(const float* __restrict__ x,
                                                      const __bf16* __restrict__ Ut,
                                                      float* __restrict__ pre) {
    const int m0   = blockIdx.x * 16;
    const int lane = threadIdx.x;
    const int half = lane >> 4;                     // 0: lanes 0-15, 1: lanes 16-31
    const int lq   = lane & 15;

    const int kBeg = blockIdx.y * (N_DIM / KSPLIT);
    const int kEnd = kBeg + (N_DIM / KSPLIT);

    const float* __restrict__ xrow = x + (size_t)(m0 + lq) * N_DIM;

    v8f c[8] = {};
    for (int kb = kBeg; kb < kEnd; kb += 32) {
        __builtin_prefetch(xrow + kb + 512, 0, 1);  // stream-ahead on x

        // A fragment (16x32 bf16): lane = row M=lq;
        // half 0 -> K = kb+0..15; half 1 -> K = kb+16..31
        v16bf a;
        const int ka = kb + half * 16;
        #pragma unroll
        for (int i = 0; i < 16; ++i)
            a[i] = (__bf16)xrow[ka + i];

        // B fragments: lane = col n; half 0 -> K=kb+0..15, half 1 -> K=kb+16..31.
        // Ut[n][k] is contiguous in k -> one 32B v16bf load per fragment.
        const int kbb = kb + half * 16;
        #pragma unroll
        for (int nt = 0; nt < 8; ++nt) {
            const v16bf b = *(const v16bf*)(Ut + (size_t)(nt * 16 + lq) * N_DIM + kbb);
            c[nt] = __builtin_amdgcn_wmma_f32_16x16x32_bf16(false, a, false, b,
                                                            (short)0, c[nt], false, false);
        }
    }

    // C layout: lane -> col lq, vgpr r -> row m0 + r + 8*half
    #pragma unroll
    for (int nt = 0; nt < 8; ++nt) {
        #pragma unroll
        for (int r = 0; r < 8; ++r) {
            const int m = m0 + r + half * 8;
            unsafeAtomicAdd(&pre[(size_t)m * R_DIM + (nt * 16 + lq)], c[nt][r]);
        }
    }
}

// ---------------- out = mask ⊙ (pre @ V^T) ----------------
// One wave per 16x64 output strip (4 n-tiles share one A fragment).
// K = 128 fully unrolled: 4 k-steps x 4 WMMAs. B loads are direct bf16.

__global__ void __launch_bounds__(32) gemm_out_kernel(const float* __restrict__ pre,
                                                      const __bf16* __restrict__ Vb,
                                                      const unsigned* __restrict__ mask,
                                                      float* __restrict__ out) {
    const int n0   = blockIdx.x * (16 * NT_OUT);
    const int m0   = blockIdx.y * 16;
    const int lane = threadIdx.x;
    const int half = lane >> 4;
    const int lq   = lane & 15;

    const float* __restrict__ arow = pre + (size_t)(m0 + lq) * R_DIM;

    v8f c[NT_OUT] = {};
    #pragma unroll
    for (int kb = 0; kb < R_DIM; kb += 32) {
        v16bf a;
        const int ka = kb + half * 16;
        #pragma unroll
        for (int i = 0; i < 16; ++i)
            a[i] = (__bf16)arow[ka + i];

        const int kbb = kb + half * 16;
        #pragma unroll
        for (int nt = 0; nt < NT_OUT; ++nt) {
            // B[k][j] = V[j][k]; Vb row j contiguous in k -> 32B v16bf load
            const v16bf b = *(const v16bf*)(Vb + (size_t)(n0 + nt * 16 + lq) * R_DIM + kbb);
            c[nt] = __builtin_amdgcn_wmma_f32_16x16x32_bf16(false, a, false, b,
                                                            (short)0, c[nt], false, false);
        }
    }

    #pragma unroll
    for (int nt = 0; nt < NT_OUT; ++nt) {
        const int   j  = n0 + nt * 16 + lq;
        const float mj = mask[j] ? 1.0f : 0.0f;
        #pragma unroll
        for (int r = 0; r < 8; ++r) {
            const int m = m0 + r + half * 8;
            out[(size_t)m * N_DIM + j] = c[nt][r] * mj;  // 64B contiguous per half-wave
        }
    }
}

// ---------------- launch ----------------

extern "C" void kernel_launch(void* const* d_in, const int* in_sizes, int n_in,
                              void* d_out, int out_size, void* d_ws, size_t ws_size,
                              hipStream_t stream) {
    const float* x       = (const float*)d_in[0];   // [512, 16384]
    const float* U       = (const float*)d_in[1];   // [16384, 128]
    const float* V       = (const float*)d_in[2];   // [16384, 128]
    // d_in[3] = indptr (unused by the reference semantics)
    const int*   indices = (const int*)d_in[4];     // [NNZ]
    const int    nnz     = in_sizes[4];

    float* out = (float*)d_out;                     // [512, 16384]

    // ws layout: pre f32 | mask u32 | Ut bf16 | Vbf bf16
    char* wsb = (char*)d_ws;
    float*    pre  = (float*)wsb;                                        wsb += (size_t)B_DIM * R_DIM * sizeof(float);
    unsigned* mask = (unsigned*)wsb;                                     wsb += (size_t)N_DIM * sizeof(unsigned);
    __bf16*   Ut   = (__bf16*)wsb;                                       wsb += (size_t)R_DIM * N_DIM * sizeof(__bf16);
    __bf16*   Vbf  = (__bf16*)wsb;

    // 1) zero pre + mask (contiguous region at start of d_ws)
    const int zwords = B_DIM * R_DIM + N_DIM;
    zero_ws_kernel<<<(zwords + 255) / 256, 256, 0, stream>>>((float*)d_ws, zwords);

    // 2) scatter column mask
    build_mask_kernel<<<(nnz + 255) / 256, 256, 0, stream>>>(indices, mask, nnz);

    // 3) U -> Ut (bf16 transpose), V -> Vbf (bf16 copy)
    dim3 gt(N_DIM / TK, R_DIM / 16);
    transpose_U_kernel<<<gt, 256, 0, stream>>>(U, Ut);
    const int ve = N_DIM * R_DIM;
    cvt_V_kernel<<<(ve + 255) / 256, 256, 0, stream>>>(V, Vbf, ve);

    // 4) pre = x @ U   (32 row-blocks x 16 k-slices, 8 WMMAs/k-step per wave)
    dim3 g1(B_DIM / 16, KSPLIT);
    gemm_pre_kernel<<<g1, 32, 0, stream>>>(x, Ut, pre);

    // 5) out = mask ⊙ (pre @ V^T)   (256 x 32 strips of 16x64)
    dim3 g2(N_DIM / (16 * NT_OUT), B_DIM / 16);
    gemm_out_kernel<<<g2, 32, 0, stream>>>(pre, Vbf, mask, out);
}
